// MultiheadAttentionWithRoPE_34239479284091
// MI455X (gfx1250) — matrix operand order repaired and checked by reading, not compiled, not run on hardware
//
#include <hip/hip_runtime.h>
#include <hip/hip_bf16.h>

typedef __attribute__((ext_vector_type(16))) __bf16 v16bf;
typedef __attribute__((ext_vector_type(8)))  __bf16 v8bf;
typedef __attribute__((ext_vector_type(8)))  float  v8f;
typedef __attribute__((ext_vector_type(4)))  float  v4f;
typedef __attribute__((ext_vector_type(8)))  unsigned short v8us;

static constexpr int Bc  = 2;
static constexpr int Tc  = 2048;
static constexpr int Dc  = 1024;
static constexpr int Hc  = 16;
static constexpr int HDc = 64;
static constexpr int BT  = Bc * Tc;
#define ROPE_SCALE 0.125f   /* 1/sqrt(64) */

// ---------------- helpers ----------------

__device__ __forceinline__ unsigned short f32_to_bf16_bits(float f) {
  unsigned u = __builtin_bit_cast(unsigned, f);
  u += 0x7FFFu + ((u >> 16) & 1u);           // round-to-nearest-even
  return (unsigned short)(u >> 16);
}
__device__ __forceinline__ v8bf ldbf8(const unsigned short* p) {
  return *reinterpret_cast<const v8bf*>(p);   // 16-byte load
}
__device__ __forceinline__ v16bf cat16(v8bf lo, v8bf hi) {
  return __builtin_shufflevector(lo, hi, 0,1,2,3,4,5,6,7,8,9,10,11,12,13,14,15);
}
__device__ __forceinline__ v16bf ldbf16(const unsigned short* p) {
  return cat16(ldbf8(p), ldbf8(p + 8));       // 32 contiguous bytes
}
__device__ __forceinline__ v8f wmma_bf16(v16bf a, v16bf b, v8f c) {
  return __builtin_amdgcn_wmma_f32_16x16x32_bf16(false, a, false, b, (short)0, c,
                                                 false, false);
}

// ------------- kernel 0: f32 -> bf16 elementwise cast (8 elems/thread) ------
__global__ __launch_bounds__(256)
void cast_kernel(const float* __restrict__ X, unsigned short* __restrict__ Y) {
  const size_t i = ((size_t)blockIdx.x * blockDim.x + threadIdx.x) * 8;
  v4f a = *reinterpret_cast<const v4f*>(X + i);
  v4f b = *reinterpret_cast<const v4f*>(X + i + 4);
  v8us o;
#pragma unroll
  for (int e = 0; e < 4; ++e) {
    o[e]     = f32_to_bf16_bits(a[e]);
    o[4 + e] = f32_to_bf16_bits(b[e]);
  }
  *reinterpret_cast<v8us*>(Y + i) = o;       // 16-byte store
}

// ------------- kernel 1: weight transpose + cast to bf16 -------------
__global__ void wt_kernel(const float* __restrict__ W,
                          unsigned short* __restrict__ Wt) {
  int k = blockIdx.x * 16 + threadIdx.x;
  int n = blockIdx.y * 16 + threadIdx.y;
  Wt[(size_t)n * Dc + k] = f32_to_bf16_bits(W[(size_t)k * Dc + n]);
}

// ------------- kernel 2: projection GEMM + bias + (RoPE) -------------
// One wave computes a 64-row x 64-col (one head) tile: 4 M-subtiles share
// every B fragment (4x weight-traffic reuse vs 16-row tiles).
// mode: 0 = Q (RoPE + scale, row-major out), 1 = K (RoPE, row-major out),
//       2 = V (no RoPE, transposed out: [(b*H+h)*HD+hd][s])
__global__ __launch_bounds__(32)
void proj_kernel(const unsigned short* __restrict__ Xb,
                 const unsigned short* __restrict__ Wt,
                 const float* __restrict__ bias,
                 unsigned short* __restrict__ dst, int mode) {
  const int lane = threadIdx.x;
  const int half = lane >> 4;
  const int n    = lane & 15;
  const int mbase = blockIdx.x * 64;
  const int h     = blockIdx.y;

  v8f acc[4][4];                         // [mi][j]
#pragma unroll
  for (int mi = 0; mi < 4; ++mi)
#pragma unroll
    for (int j = 0; j < 4; ++j) acc[mi][j] = (v8f)0.0f;

  const unsigned short* xrow[4];
#pragma unroll
  for (int mi = 0; mi < 4; ++mi)
    xrow[mi] = Xb + (size_t)(mbase + mi * 16 + n) * Dc;  // A: M = lane%16
  const unsigned short* brow[4];
#pragma unroll
  for (int j = 0; j < 4; ++j)
    brow[j] = Wt + (size_t)(h * 64 + j * 16 + n) * Dc;   // B: N = lane%16

  for (int kk = 0; kk < Dc; kk += 32) {
    const int k0 = kk + 16 * half;       // A: K = 16*half+e (matches B)
    const int k1 = kk + 16 * half + 8;
    v16bf a[4];
#pragma unroll
    for (int mi = 0; mi < 4; ++mi)
      a[mi] = cat16(ldbf8(xrow[mi] + k0), ldbf8(xrow[mi] + k1));
#pragma unroll
    for (int j = 0; j < 4; ++j) {
      v16bf b = ldbf16(brow[j] + kk + 16 * half);        // B: K = 16*half+e
#pragma unroll
      for (int mi = 0; mi < 4; ++mi)
        acc[mi][j] = wmma_bf16(a[mi], b, acc[mi][j]);
    }
  }

  float bj[4];
#pragma unroll
  for (int j = 0; j < 4; ++j) bj[j] = bias[h * 64 + j * 16 + n];

  const float iv0 = __powf(10000.0f, -((float)n)        / 32.0f);  // j even
  const float iv1 = __powf(10000.0f, -((float)(16 + n)) / 32.0f);  // j odd

#pragma unroll
  for (int mi = 0; mi < 4; ++mi) {
    float y[4][8];
    if (mode < 2) {
      // RoPE: hd = 16*j + n ; rotate-half partner tile is j^2, same lane/slot.
#pragma unroll
      for (int r = 0; r < 8; ++r) {
        const int row = mbase + mi * 16 + r + 8 * half;  // C: M = r + 8*half
        const float pos = (float)(row & (Tc - 1));
        float s0, c0, s1, c1;
        __sincosf(pos * iv0, &s0, &c0);
        __sincosf(pos * iv1, &s1, &c1);
#pragma unroll
        for (int j = 0; j < 4; ++j) {
          const float cs = (j & 1) ? c1 : c0;
          const float sn = (j & 1) ? s1 : s0;
          const float sign = (j < 2) ? -1.0f : 1.0f;
          float v = (acc[mi][j][r] + bj[j]) * cs +
                    sign * (acc[mi][j ^ 2][r] + bj[j ^ 2]) * sn;
          if (mode == 0) v *= ROPE_SCALE;
          y[j][r] = v;
        }
      }
    } else {
#pragma unroll
      for (int r = 0; r < 8; ++r)
#pragma unroll
        for (int j = 0; j < 4; ++j) y[j][r] = acc[mi][j][r] + bj[j];
    }

    if (mode == 2) {  // V transposed: hd-major so PV B-frags are contiguous
#pragma unroll
      for (int r = 0; r < 8; ++r) {
        const int row = mbase + mi * 16 + r + 8 * half;
        const int b = row >> 11, s = row & (Tc - 1);
#pragma unroll
        for (int j = 0; j < 4; ++j) {
          const int hd = j * 16 + n;
          dst[((size_t)((b * Hc + h) * HDc + hd)) * Tc + s] =
              f32_to_bf16_bits(y[j][r]);
        }
      }
    } else {
#pragma unroll
      for (int r = 0; r < 8; ++r) {
        const int row = mbase + mi * 16 + r + 8 * half;
#pragma unroll
        for (int j = 0; j < 4; ++j)
          dst[(size_t)row * Dc + h * 64 + j * 16 + n] = f32_to_bf16_bits(y[j][r]);
      }
    }
  }
}

// ------------- kernel 3: flash attention (one wave per b,h,32 q-rows) -------
// Two 16-row M-subtiles share every K/V fragment (halves K/V L2 traffic).
__global__ __launch_bounds__(32)
void attn_kernel(const unsigned short* __restrict__ Qb,
                 const unsigned short* __restrict__ Kb,
                 const unsigned short* __restrict__ Vt,
                 unsigned short* __restrict__ Ab) {
  __shared__ unsigned short lds[32 * 40];  // 32 x 32 P tile, row stride 40 (pad)
  const int lane = threadIdx.x, half = lane >> 4, n = lane & 15;
  const int qbase = blockIdx.x * 32;
  const int h = blockIdx.y;
  const int b = blockIdx.z;

  // Q fragments: [mi][ks]; Q already RoPE'd + scaled.
  v16bf aq[2][2];
#pragma unroll
  for (int mi = 0; mi < 2; ++mi) {
    const unsigned short* qrow =
        Qb + (size_t)(b * Tc + qbase + mi * 16 + n) * Dc + h * 64;
#pragma unroll
    for (int ks = 0; ks < 2; ++ks)
      aq[mi][ks] = cat16(ldbf8(qrow + ks * 32 + 16 * half),
                         ldbf8(qrow + ks * 32 + 16 * half + 8));
  }

  float m[2][8], l[2][8];
  v8f o[2][4];
#pragma unroll
  for (int mi = 0; mi < 2; ++mi) {
#pragma unroll
    for (int r = 0; r < 8; ++r) { m[mi][r] = -1e30f; l[mi][r] = 0.0f; }
#pragma unroll
    for (int j = 0; j < 4; ++j) o[mi][j] = (v8f)0.0f;
  }

  const unsigned short* kbase = Kb + (size_t)(b * Tc) * Dc + h * 64;
  const unsigned short* vbase = Vt + (size_t)((b * Hc + h) * HDc) * Tc;

  for (int s0 = 0; s0 < Tc; s0 += 32) {
    // ---- scores: 2 M-subtiles x 2 N-tiles over HD=64 ----
    v8f sc[2][2];
#pragma unroll
    for (int mi = 0; mi < 2; ++mi)
#pragma unroll
      for (int j = 0; j < 2; ++j) sc[mi][j] = (v8f)0.0f;
#pragma unroll
    for (int j = 0; j < 2; ++j) {
      const unsigned short* krow = kbase + (size_t)(s0 + j * 16 + n) * Dc;
#pragma unroll
      for (int ks = 0; ks < 2; ++ks) {
        v16bf bk = ldbf16(krow + ks * 32 + 16 * half);
#pragma unroll
        for (int mi = 0; mi < 2; ++mi)
          sc[mi][j] = wmma_bf16(aq[mi][ks], bk, sc[mi][j]);
      }
    }
    // prefetch next chunk's K/V rows (global_prefetch_b8) during softmax
    if (s0 + 32 < Tc) {
#pragma unroll
      for (int j = 0; j < 2; ++j)
        __builtin_prefetch(kbase + (size_t)(s0 + 32 + j * 16 + n) * Dc, 0, 1);
#pragma unroll
      for (int j = 0; j < 4; ++j)
        __builtin_prefetch(vbase + (size_t)(j * 16 + n) * Tc + s0 + 32, 0, 1);
    }
    // ---- online softmax (row-wise across the 16 lanes sharing a row) ----
#pragma unroll
    for (int mi = 0; mi < 2; ++mi) {
      float alpha[8];
#pragma unroll
      for (int r = 0; r < 8; ++r) {
        float v = fmaxf(sc[mi][0][r], sc[mi][1][r]);
#pragma unroll
        for (int d = 1; d < 16; d <<= 1) v = fmaxf(v, __shfl_xor(v, d, 16));
        const float mnew = fmaxf(m[mi][r], v);
        alpha[r] = __expf(m[mi][r] - mnew);
        m[mi][r] = mnew;
      }
#pragma unroll
      for (int r = 0; r < 8; ++r) {
        float p0 = __expf(sc[mi][0][r] - m[mi][r]);
        float p1 = __expf(sc[mi][1][r] - m[mi][r]);
        sc[mi][0][r] = p0; sc[mi][1][r] = p1;
        float v = p0 + p1;
#pragma unroll
        for (int d = 1; d < 16; d <<= 1) v += __shfl_xor(v, d, 16);
        l[mi][r] = l[mi][r] * alpha[r] + v;
#pragma unroll
        for (int j = 0; j < 4; ++j) o[mi][j][r] = o[mi][j][r] * alpha[r];
      }
      // write this subtile's P into LDS (C-layout -> A-layout transpose)
#pragma unroll
      for (int r = 0; r < 8; ++r) {
        const int row = mi * 16 + r + 8 * half;
        lds[row * 40 + n]      = f32_to_bf16_bits(sc[mi][0][r]);
        lds[row * 40 + 16 + n] = f32_to_bf16_bits(sc[mi][1][r]);
      }
    }
    __syncthreads();
    v16bf pa[2];
#pragma unroll
    for (int mi = 0; mi < 2; ++mi) {
      const unsigned short* pr = lds + (mi * 16 + n) * 40;
      pa[mi] = cat16(ldbf8(pr + 16 * half), ldbf8(pr + 16 * half + 8));
    }
    // ---- P (32x32) x V (32x16 per hd-tile); V frag shared by both mi ----
#pragma unroll
    for (int j = 0; j < 4; ++j) {
      v16bf bv = ldbf16(vbase + (size_t)(j * 16 + n) * Tc + s0 + 16 * half);
#pragma unroll
      for (int mi = 0; mi < 2; ++mi)
        o[mi][j] = wmma_bf16(pa[mi], bv, o[mi][j]);
    }
    __syncthreads();
  }
  // ---- normalize + store (row-major bf16 for final GEMM) ----
#pragma unroll
  for (int mi = 0; mi < 2; ++mi)
#pragma unroll
    for (int r = 0; r < 8; ++r) {
      const float inv = 1.0f / l[mi][r];
      const int row = b * Tc + qbase + mi * 16 + r + 8 * half;
#pragma unroll
      for (int j = 0; j < 4; ++j)
        Ab[(size_t)row * Dc + h * 64 + j * 16 + n] =
            f32_to_bf16_bits(o[mi][j][r] * inv);
    }
}

// ------------- kernel 4: output projection (bf16 GEMM -> f32 + bias) -------
// 64-row x 64-col tile per wave, B fragments shared by 4 M-subtiles.
__global__ __launch_bounds__(32)
void oproj_kernel(const unsigned short* __restrict__ A,
                  const unsigned short* __restrict__ Wt,
                  const float* __restrict__ bias,
                  float* __restrict__ out) {
  const int lane = threadIdx.x, half = lane >> 4, n = lane & 15;
  const int mbase = blockIdx.x * 64;
  const int g = blockIdx.y;   // 64-column group
  v8f acc[4][4];
#pragma unroll
  for (int mi = 0; mi < 4; ++mi)
#pragma unroll
    for (int j = 0; j < 4; ++j) acc[mi][j] = (v8f)0.0f;
  const unsigned short* arow[4];
#pragma unroll
  for (int mi = 0; mi < 4; ++mi)
    arow[mi] = A + (size_t)(mbase + mi * 16 + n) * Dc;
  const unsigned short* brow[4];
#pragma unroll
  for (int j = 0; j < 4; ++j)
    brow[j] = Wt + (size_t)(g * 64 + j * 16 + n) * Dc;
  for (int kk = 0; kk < Dc; kk += 32) {
    const int k0 = kk + 16 * half;
    const int k1 = kk + 16 * half + 8;
    v16bf a[4];
#pragma unroll
    for (int mi = 0; mi < 4; ++mi)
      a[mi] = cat16(ldbf8(arow[mi] + k0), ldbf8(arow[mi] + k1));
#pragma unroll
    for (int j = 0; j < 4; ++j) {
      v16bf bm = ldbf16(brow[j] + kk + 16 * half);
#pragma unroll
      for (int mi = 0; mi < 4; ++mi)
        acc[mi][j] = wmma_bf16(a[mi], bm, acc[mi][j]);
    }
  }
#pragma unroll
  for (int mi = 0; mi < 4; ++mi)
#pragma unroll
    for (int r = 0; r < 8; ++r) {
      const int row = mbase + mi * 16 + r + 8 * half;
#pragma unroll
      for (int j = 0; j < 4; ++j) {
        const int col = g * 64 + j * 16 + n;
        out[(size_t)row * Dc + col] = acc[mi][j][r] + bias[col];
      }
    }
}

// ------------------------- launch -------------------------
extern "C" void kernel_launch(void* const* d_in, const int* in_sizes, int n_in,
                              void* d_out, int out_size, void* d_ws, size_t ws_size,
                              hipStream_t stream) {
  const float* q  = (const float*)d_in[0];
  const float* k  = (const float*)d_in[1];
  const float* v  = (const float*)d_in[2];
  const float* Wq = (const float*)d_in[3];
  const float* bq = (const float*)d_in[4];
  const float* Wk = (const float*)d_in[5];
  const float* bk = (const float*)d_in[6];
  const float* Wv = (const float*)d_in[7];
  const float* bv = (const float*)d_in[8];
  const float* Wo = (const float*)d_in[9];
  const float* bo = (const float*)d_in[10];
  float* out = (float*)d_out;

  unsigned short* ws = (unsigned short*)d_ws;
  const size_t DD  = (size_t)Dc * Dc;   // 1M elems
  const size_t BTD = (size_t)BT * Dc;   // 4M elems
  unsigned short* WqT = ws;
  unsigned short* WkT = WqT + DD;
  unsigned short* WvT = WkT + DD;
  unsigned short* WoT = WvT + DD;
  unsigned short* Xq  = WoT + DD;
  unsigned short* Xk  = Xq + BTD;
  unsigned short* Xv  = Xk + BTD;
  unsigned short* Qb  = Xv + BTD;
  unsigned short* Kb  = Qb + BTD;
  unsigned short* VtB = Kb + BTD;
  unsigned short* Ab  = VtB + BTD;      // total 64 MB

  // stage 0: cast inputs to bf16 (each element converted exactly once)
  const int castBlocks = (int)(BTD / (256 * 8));
  cast_kernel<<<castBlocks, 256, 0, stream>>>(q, Xq);
  cast_kernel<<<castBlocks, 256, 0, stream>>>(k, Xk);
  cast_kernel<<<castBlocks, 256, 0, stream>>>(v, Xv);

  // stage 1: transpose+cast weights
  dim3 tb(16, 16), tg(Dc / 16, Dc / 16);
  wt_kernel<<<tg, tb, 0, stream>>>(Wq, WqT);
  wt_kernel<<<tg, tb, 0, stream>>>(Wk, WkT);
  wt_kernel<<<tg, tb, 0, stream>>>(Wv, WvT);
  wt_kernel<<<tg, tb, 0, stream>>>(Wo, WoT);

  // stage 2: Q/K/V projections (+bias, +RoPE, Q scaled, V transposed)
  dim3 pg(BT / 64, Hc);
  proj_kernel<<<pg, 32, 0, stream>>>(Xq, WqT, bq, Qb, 0);
  proj_kernel<<<pg, 32, 0, stream>>>(Xk, WkT, bk, Kb, 1);
  proj_kernel<<<pg, 32, 0, stream>>>(Xv, WvT, bv, VtB, 2);

  // stage 3: flash attention (32 q-rows per wave)
  dim3 ag(Tc / 32, Hc, Bc);
  attn_kernel<<<ag, 32, 0, stream>>>(Qb, Kb, VtB, Ab);

  // stage 4: output projection
  oproj_kernel<<<pg, 32, 0, stream>>>(Ab, WoT, bo, out);
}
